// SpikingSelfAttention_26250840113225
// MI455X (gfx1250) — matrix operand order repaired and checked by reading, not compiled, not run on hardware
//
#include <hip/hip_runtime.h>
#include <hip/hip_bf16.h>

typedef __attribute__((ext_vector_type(16))) _Float16 v16h;
typedef __attribute__((ext_vector_type(8)))  _Float16 v8h;
typedef __attribute__((ext_vector_type(8)))  float    v8f;

#define S_LEN 2048
#define H_DIM 1024
#define NH 16
#define HD 64
#define WIN 64

// ---------------------------------------------------------------------------
// f32 -> f16 conversion
// ---------------------------------------------------------------------------
__global__ void cvt_f32_to_f16_kernel(const float* __restrict__ src,
                                      _Float16* __restrict__ dst, int n) {
    int i = blockIdx.x * blockDim.x + threadIdx.x;
    if (i < n) dst[i] = (_Float16)src[i];
}

// ---------------------------------------------------------------------------
// GEMM: out = spike(X[4096,1024] @ W[1024,1024]^T).
// Register-blocked: each wave owns a 32x64 C tile = 2 A-frags x 4 B-frags
// -> 8 WMMAs per K-step. MODE is a template arg so each instantiation gets a
// straight-line, branch-free epilogue.
// MODE 0/1: write f16 spikes head-major  [b][h][s][d]   (Q / K buffers)
// MODE 2  : write f16 spikes transposed  [b][h][d][s]   (V buffer)
// MODE 3  : write f32 spikes flat        [row][col]     (final output)
// ---------------------------------------------------------------------------
template <int MODE>
__global__ __launch_bounds__(128)
void gemm_spike_kernel(const _Float16* __restrict__ X,
                       const _Float16* __restrict__ W,
                       void* __restrict__ out) {
    const int K = H_DIM;
    const int wave = threadIdx.x >> 5;
    const int lane = threadIdx.x & 31;
    const int hf   = lane >> 4;      // half-wave select
    const int lrow = lane & 15;
    const int m0 = blockIdx.x * 32;                 // grid.x = 128 (M = 4096)
    const int n0 = (blockIdx.y * 4 + wave) * 64;    // grid.y = 4   (N = 1024)

    const _Float16* xr[2];
    xr[0] = X + (size_t)(m0 + lrow) * K;
    xr[1] = X + (size_t)(m0 + 16 + lrow) * K;
    const _Float16* wr[4];
#pragma unroll
    for (int j = 0; j < 4; j++) wr[j] = W + (size_t)(n0 + j * 16 + lrow) * K;

    v8f c[2][4] = {};
    for (int kk = 0; kk < K; kk += 32) {
        // A fragments: row lrow (+16), elems 0-15 = K kk+hf*16 .. +15
        v16h a[2];
#pragma unroll
        for (int mi = 0; mi < 2; mi++) {
            v8h lo = *(const v8h*)(xr[mi] + kk + hf * 16);
            v8h hi = *(const v8h*)(xr[mi] + kk + hf * 16 + 8);
#pragma unroll
            for (int i = 0; i < 8; i++) { a[mi][i] = lo[i]; a[mi][8 + i] = hi[i]; }
        }
        // B fragments: column lrow of W^T = row of W, K kk+hf*16 .. +15
        v16h bm[4];
#pragma unroll
        for (int j = 0; j < 4; j++)
            bm[j] = *(const v16h*)(wr[j] + kk + hf * 16);
#pragma unroll
        for (int mi = 0; mi < 2; mi++)
#pragma unroll
            for (int j = 0; j < 4; j++)
                c[mi][j] = __builtin_amdgcn_wmma_f32_16x16x32_f16(
                    false, a[mi], false, bm[j], (short)0, c[mi][j], false, false);
    }

    // Branch-free epilogue (MODE known at compile time).
#pragma unroll
    for (int mi = 0; mi < 2; mi++) {
        const int row0 = m0 + mi * 16 + hf * 8;
#pragma unroll
        for (int j = 0; j < 4; j++) {
            const int col = n0 + j * 16 + lrow;
#pragma unroll
            for (int i = 0; i < 8; i++) {
                const int row = row0 + i;
                const float s = (c[mi][j][i] >= 1.0f) ? 1.0f : 0.0f;
                if (MODE == 3) {
                    ((float*)out)[(size_t)row * H_DIM + col] = s;
                } else {
                    const int bb = row >> 11, ss = row & (S_LEN - 1);
                    const int hh = col >> 6,  dd = col & (HD - 1);
                    _Float16* o = (_Float16*)out;
                    if (MODE == 2)
                        o[(((size_t)(bb * NH + hh) * HD + dd) * S_LEN) + ss] = (_Float16)s;
                    else
                        o[(((size_t)(bb * NH + hh) * S_LEN + ss) * HD) + dd] = (_Float16)s;
                }
            }
        }
    }
}

// ---------------------------------------------------------------------------
// Banded attention, one wave per (b, head, 16-query tile).
// scores = Q K^T / 8 over 9 key tiles, band mask |i-j|<=64, f32 softmax,
// attn = P V  (V stored transposed so B fragments are contiguous).
// ---------------------------------------------------------------------------
__global__ __launch_bounds__(32)
void banded_attn_kernel(const _Float16* __restrict__ Q,
                        const _Float16* __restrict__ Ks,
                        const _Float16* __restrict__ Vt,
                        _Float16* __restrict__ attn) {
    __shared__ __attribute__((aligned(16))) float    sc[16][144];
    __shared__ __attribute__((aligned(16))) _Float16 sp[16][160];

    const int qt = blockIdx.x;   // 0..127
    const int h  = blockIdx.y;   // 0..15
    const int b  = blockIdx.z;   // 0..1
    const int q0 = qt * 16;
    const int lane = threadIdx.x;
    const int hf   = lane >> 4;
    const int lrow = lane & 15;

    const _Float16* qbase = Q  + (size_t)(b * NH + h) * S_LEN * HD;
    const _Float16* kbase = Ks + (size_t)(b * NH + h) * S_LEN * HD;
    const _Float16* vbase = Vt + (size_t)(b * NH + h) * HD * S_LEN;

    // Q fragments for K-dim = 0..31 and 32..63 (loaded once)
    v16h aq[2];
    const _Float16* qr = qbase + (size_t)(q0 + lrow) * HD;
#pragma unroll
    for (int ks = 0; ks < 2; ks++) {
        v8h lo = *(const v8h*)(qr + ks * 32 + hf * 16);
        v8h hi = *(const v8h*)(qr + ks * 32 + hf * 16 + 8);
#pragma unroll
        for (int i = 0; i < 8; i++) { aq[ks][i] = lo[i]; aq[ks][8 + i] = hi[i]; }
    }

    // ---- scores over 9 key tiles -----------------------------------------
    for (int t = 0; t < 9; t++) {
        int kb  = q0 - WIN + t * 16;
        int key = kb + lrow;                  // column this lane owns
        bool ok = (key >= 0) && (key < S_LEN);
        v16h b0 = {}, b1 = {};
        if (ok) {
            const _Float16* kr = kbase + (size_t)key * HD;
            b0 = *(const v16h*)(kr + hf * 16);
            b1 = *(const v16h*)(kr + 32 + hf * 16);
        }
        v8f c = {};
        c = __builtin_amdgcn_wmma_f32_16x16x32_f16(false, aq[0], false, b0,
                                                   (short)0, c, false, false);
        c = __builtin_amdgcn_wmma_f32_16x16x32_f16(false, aq[1], false, b1,
                                                   (short)0, c, false, false);
#pragma unroll
        for (int i = 0; i < 8; i++) {
            int m  = hf * 8 + i;
            int dq = (q0 + m) - key;
            bool band = ok && (dq <= WIN) && (dq >= -WIN);
            sc[m][t * 16 + lrow] = band ? c[i] * 0.125f : -1e9f;
        }
    }
    __syncthreads();

    // ---- f32 softmax over the 144-wide band (one row per thread 0..15) ---
    if (hf == 0) {
        float mx = -1e30f;
        for (int j = 0; j < 144; j++) mx = fmaxf(mx, sc[lrow][j]);
        float sum = 0.0f;
        for (int j = 0; j < 144; j++) {
            float e = __expf(sc[lrow][j] - mx);
            sc[lrow][j] = e;
            sum += e;
        }
        float inv = 1.0f / sum;
        for (int j = 0; j < 144; j++) sp[lrow][j] = (_Float16)(sc[lrow][j] * inv);
        for (int j = 144; j < 160; j++) sp[lrow][j] = (_Float16)0.0f;
    }
    __syncthreads();

    // ---- P @ V : K = 160 (padded), N = 64 --------------------------------
    const int kb0 = q0 - WIN;
    for (int nt = 0; nt < 4; nt++) {
        v8f c = {};
        for (int ks = 0; ks < 5; ks++) {
            v8h lo = *(const v8h*)(&sp[lrow][ks * 32 + hf * 16]);
            v8h hi = *(const v8h*)(&sp[lrow][ks * 32 + hf * 16 + 8]);
            v16h a;
#pragma unroll
            for (int i = 0; i < 8; i++) { a[i] = lo[i]; a[8 + i] = hi[i]; }

            int s0 = kb0 + ks * 32 + hf * 16;   // 16-aligned key segment
            v16h bm = {};
            if (s0 >= 0 && s0 + 16 <= S_LEN)
                bm = *(const v16h*)(vbase + (size_t)(nt * 16 + lrow) * S_LEN + s0);
            c = __builtin_amdgcn_wmma_f32_16x16x32_f16(false, a, false, bm,
                                                       (short)0, c, false, false);
        }
#pragma unroll
        for (int i = 0; i < 8; i++) {
            int m    = hf * 8 + i;
            int srow = q0 + m;
            int col  = h * HD + nt * 16 + lrow;
            attn[(size_t)(b * S_LEN + srow) * H_DIM + col] = (_Float16)c[i];
        }
    }
}

// ---------------------------------------------------------------------------
// Workspace layout (all _Float16 elements from ws base):
//   XH @ 0, WQ/WK/WV/WO (1M each), QB/KB head-major, VT transposed, AT attn.
// total 25,165,824 halves = 48 MB
// ---------------------------------------------------------------------------
extern "C" void kernel_launch(void* const* d_in, const int* in_sizes, int n_in,
                              void* d_out, int out_size, void* d_ws, size_t ws_size,
                              hipStream_t stream) {
    const float* s_t = (const float*)d_in[0];
    const float* Wq  = (const float*)d_in[1];
    const float* Wk  = (const float*)d_in[2];
    const float* Wv  = (const float*)d_in[3];
    const float* Wo  = (const float*)d_in[4];
    float* out = (float*)d_out;

    _Float16* ws = (_Float16*)d_ws;
    _Float16* XH = ws;
    _Float16* WQ = ws + 4194304;
    _Float16* WK = ws + 5242880;
    _Float16* WV = ws + 6291456;
    _Float16* WO = ws + 7340032;
    _Float16* QB = ws + 8388608;
    _Float16* KB = ws + 12582912;
    _Float16* VT = ws + 16777216;
    _Float16* AT = ws + 20971520;

    // 1) convert inputs to f16
    {
        int n = 2 * S_LEN * H_DIM;  // 4,194,304
        cvt_f32_to_f16_kernel<<<(n + 255) / 256, 256, 0, stream>>>(s_t, XH, n);
        int nw = H_DIM * H_DIM;     // 1,048,576
        cvt_f32_to_f16_kernel<<<(nw + 255) / 256, 256, 0, stream>>>(Wq, WQ, nw);
        cvt_f32_to_f16_kernel<<<(nw + 255) / 256, 256, 0, stream>>>(Wk, WK, nw);
        cvt_f32_to_f16_kernel<<<(nw + 255) / 256, 256, 0, stream>>>(Wv, WV, nw);
        cvt_f32_to_f16_kernel<<<(nw + 255) / 256, 256, 0, stream>>>(Wo, WO, nw);
    }

    // 2) Q/K/V projections + spike (32x64 C tile per wave)
    dim3 ggrid(128, 4, 1);    // (M/32) x (N/256)
    dim3 gblock(128, 1, 1);   // 4 waves, 32x64 C tile each
    gemm_spike_kernel<0><<<ggrid, gblock, 0, stream>>>(XH, WQ, (void*)QB);
    gemm_spike_kernel<1><<<ggrid, gblock, 0, stream>>>(XH, WK, (void*)KB);
    gemm_spike_kernel<2><<<ggrid, gblock, 0, stream>>>(XH, WV, (void*)VT);

    // 3) banded attention
    dim3 agrid(S_LEN / 16, NH, 2);
    banded_attn_kernel<<<agrid, dim3(32, 1, 1), 0, stream>>>(QB, KB, VT, AT);

    // 4) output projection + spike -> f32 result
    gemm_spike_kernel<3><<<ggrid, gblock, 0, stream>>>(AT, WO, (void*)out);
}
